// Seq2Seq3_65360812310790
// MI455X (gfx1250) — compile-verified, matrix operand order checked by reading, not run on hardware
//
#include <hip/hip_runtime.h>

typedef __attribute__((ext_vector_type(16))) __bf16 v16bf;
typedef __attribute__((ext_vector_type(8)))  float  v8f;
typedef __attribute__((ext_vector_type(2)))  float  v2f;

// ---- problem sizes ----
#define BSZ   64
#define TIN   128
#define FIN   64
#define HH    512
#define GG    2048              // 4*H
#define TOUT  64
#define FOUT  64
#define KENC  (FIN + HH)        // 576
#define KT_ENC (KENC / 32)      // 18
#define KT_DEC ((2 * HH) / 32)  // 32
#define NTILES (GG / 16)        // 128 packed gate-col tiles (global)
#define NB    64                // persistent blocks: 4 batch-groups x 16 col-groups
#define BT    256               // threads per block (8 waves)

// K-index inside a 32-wide chunk for the bf16 A/B fragment layout.
__device__ __forceinline__ int kmap(int lane, int j) {
  return (j < 8 ? j : j + 8) + (lane >= 16 ? 8 : 0);
}

__device__ __forceinline__ float sigf(float x) {
  return 1.0f / (1.0f + __expf(-x));
}

// ------------------------------------------------------------------
// Pack kernel: fp32 -> bf16 weight fragments (gate-interleaved cols),
// x fragments, combined biases, zero h-fragment buffer 0, reset barrier.
// ------------------------------------------------------------------
__global__ void pack_kernel(const float* __restrict__ x,
                            const float* __restrict__ Wih1, const float* __restrict__ Whh1,
                            const float* __restrict__ bih1, const float* __restrict__ bhh1,
                            const float* __restrict__ Wih2, const float* __restrict__ Whh2,
                            const float* __restrict__ bih2, const float* __restrict__ bhh2,
                            __bf16* w1p, __bf16* w2p, __bf16* xfrag,
                            float* b1, float* b2, __bf16* hfrag, unsigned* bar)
{
  const long tid = (long)blockIdx.x * blockDim.x + threadIdx.x;
  const long nth = (long)gridDim.x * blockDim.x;
  if (tid < 4) bar[tid] = 0u;

  // ---- W1 fragments: [kt 0..17][nt 0..127][lane][j] ----
  for (long i = tid; i < (long)KT_ENC * NTILES * 32; i += nth) {
    int lane = (int)(i & 31); long f = i >> 5;
    int nt = (int)(f % NTILES); int kt = (int)(f / NTILES);
    int ncol = nt * 16 + (lane & 15);      // packed col = 4*h + gate
    int gsel = ncol & 3, hc = ncol >> 2;
    long srow = (long)gsel * HH + hc;      // row in (4H, K) torch weight
    __bf16* o = w1p + ((long)(kt * NTILES + nt) * 32 + lane) * 16;
    for (int j = 0; j < 16; ++j) {
      int k = kt * 32 + kmap(lane, j);
      float v = (k < FIN) ? Wih1[srow * FIN + k] : Whh1[srow * HH + (k - FIN)];
      o[j] = (__bf16)v;
    }
  }
  // ---- W2 fragments: [kt 0..31][nt][lane][j]; K = [y(512); h(512)] ----
  for (long i = tid; i < (long)KT_DEC * NTILES * 32; i += nth) {
    int lane = (int)(i & 31); long f = i >> 5;
    int nt = (int)(f % NTILES); int kt = (int)(f / NTILES);
    int ncol = nt * 16 + (lane & 15);
    int gsel = ncol & 3, hc = ncol >> 2;
    long srow = (long)gsel * HH + hc;
    __bf16* o = w2p + ((long)(kt * NTILES + nt) * 32 + lane) * 16;
    for (int j = 0; j < 16; ++j) {
      int k = kt * 32 + kmap(lane, j);
      float v = (k < HH) ? Wih2[srow * HH + k] : Whh2[srow * HH + (k - HH)];
      o[j] = (__bf16)v;
    }
  }
  // ---- x fragments: [t][bg][kt 0..1][lane][j] ----
  for (long i = tid; i < (long)TIN * 4 * 2 * 32; i += nth) {
    int lane = (int)(i & 31); long f = i >> 5;
    int kt = (int)(f & 1); long r = f >> 1;
    int bg = (int)(r & 3); int t = (int)(r >> 2);
    int b = bg * 16 + (lane & 15);
    __bf16* o = xfrag + (((long)(t * 4 + bg) * 2 + kt) * 32 + lane) * 16;
    for (int j = 0; j < 16; ++j) {
      int k = kt * 32 + kmap(lane, j);
      o[j] = (__bf16)x[((long)b * TIN + t) * FIN + k];
    }
  }
  // ---- combined biases in packed-col order ----
  for (long i = tid; i < GG; i += nth) {
    long srow = (long)(i & 3) * HH + (i >> 2);
    b1[i] = bih1[srow] + bhh1[srow];
    b2[i] = bih2[srow] + bhh2[srow];
  }
  // ---- zero h fragment buffer 0 (h0 = 0) ----
  for (long i = tid; i < 4L * 16 * 512; i += nth) hfrag[i] = (__bf16)0.0f;
}

// ------------------------------------------------------------------
// grid-wide barrier (epoch) for the persistent kernel
// ------------------------------------------------------------------
__device__ __forceinline__ void grid_sync(unsigned* bar, unsigned& epoch)
{
  __syncthreads();
  if (threadIdx.x == 0) {
    __threadfence();
    unsigned e = epoch;
    if (atomicAdd(&bar[0], 1u) == NB - 1) {
      atomicExch(&bar[0], 0u);
      __threadfence();
      atomicAdd(&bar[1], 1u);
    } else {
      while (__hip_atomic_load(&bar[1], __ATOMIC_ACQUIRE, __HIP_MEMORY_SCOPE_AGENT) <= e) {
        __builtin_amdgcn_s_sleep(1);
      }
    }
    epoch = e + 1;
  }
  __syncthreads();
}

// ------------------------------------------------------------------
// Async global->LDS staging (16B granules, ASYNCcnt tracked).
// ------------------------------------------------------------------
__device__ __forceinline__ void stage16(const void* __restrict__ src,
                                        void* ldsdst, int chunks, int tid)
{
  for (int c = tid; c < chunks; c += BT) {
    unsigned lds = (unsigned)(size_t)((char*)ldsdst + (long)c * 16);
    unsigned long long ga = (unsigned long long)(size_t)((const char*)src + (long)c * 16);
    asm volatile("global_load_async_to_lds_b128 %0, %1, off"
                 :: "v"(lds), "v"(ga) : "memory");
  }
}
__device__ __forceinline__ void stage_wait()
{
  asm volatile("s_wait_asynccnt 0" ::: "memory");
  __syncthreads();
}

// ------------------------------------------------------------------
// LSTM cell update for this block's 16x32 (batch x h) slice.
// ------------------------------------------------------------------
__device__ __forceinline__ void cell_update(const float* gates, float* ctile,
                                            __bf16* hW, float* hlin,
                                            int bg, int cg, int tid)
{
#pragma unroll
  for (int k = 0; k < 2; ++k) {
    int idx = tid + k * BT;
    int m = idx >> 5, hc = idx & 31;
    float gi = gates[m * 128 + hc * 4 + 0];
    float gf = gates[m * 128 + hc * 4 + 1];
    float gg = gates[m * 128 + hc * 4 + 2];
    float go = gates[m * 128 + hc * 4 + 3];
    float c = ctile[m * 32 + hc];
    c = sigf(gf) * c + sigf(gi) * tanhf(gg);
    float h = sigf(go) * tanhf(c);
    ctile[m * 32 + hc] = c;
    int hcg = cg * 32 + hc;                 // global h column
    int kt = hcg >> 5, kk = hcg & 31;
    int lh = (kk >> 3) & 1;
    int jj = (kk & 7) + (kk >= 16 ? 8 : 0);
    hW[((long)(bg * 16 + kt) * 32 + (lh * 16 + m)) * 16 + jj] = (__bf16)h;
    if (hlin) hlin[(long)(bg * 16 + m) * HH + hcg] = h;
  }
}

// ------------------------------------------------------------------
// Persistent seq2seq kernel.  64 blocks = 4 batch-groups x 16 col-groups.
// Weights register-resident (enc 144, dec 256 VGPRs/wave); A fragments
// async-staged to LDS (decoder y staged once); Linear folded into the
// decoder with fp32 WMMA, A from LDS, streaming Wlin once from HBM.
// ------------------------------------------------------------------
__global__ void __launch_bounds__(BT, 1)
seq2seq_kernel(const __bf16* __restrict__ w1p, const __bf16* __restrict__ w2p,
               const __bf16* __restrict__ xfrag, __bf16* hfrag, __bf16* yfrag,
               float* hlin, const float* __restrict__ b1, const float* __restrict__ b2,
               const float* __restrict__ Wlin, const float* __restrict__ blin,
               float* __restrict__ out, unsigned* bar)
{
  const int tid  = threadIdx.x;
  const int lane = tid & 31, wv = tid >> 5;
  const int bg   = blockIdx.x & 3;    // batch group (16 rows)
  const int cg   = blockIdx.x >> 2;   // col group (128 packed gate cols / 32 h cols)
  unsigned epoch = 0;

  __shared__ __bf16 AsmLDS[KT_DEC * 512];   // 32 KB: A fragments (x|y part + h part)
  __shared__ float  hlinLDS[16 * HH];       // 32 KB: fp32 h rows for the Linear
  __shared__ float  gates[16 * 128];        // 8 KB
  __shared__ float  ctile[16 * 32];         // 2 KB
  for (int i = tid; i < 16 * 32; i += BT) ctile[i] = 0.0f;

  __bf16* hbuf0 = hfrag;
  __bf16* hbuf1 = hfrag + 4L * 16 * 512;

  const int nt0 = cg * 8 + wv;              // this wave's global 16-col tile
  const int mh  = (lane >= 16) ? 8 : 0;
  const float bb1v = b1[cg * 128 + wv * 16 + (lane & 15)];
  const float bb2v = b2[cg * 128 + wv * 16 + (lane & 15)];

  // ---------------- encoder ----------------
  {
    v16bf wE[KT_ENC];
#pragma unroll
    for (int kt = 0; kt < KT_ENC; ++kt)
      wE[kt] = *(const v16bf*)(w1p + ((long)(kt * NTILES + nt0) * 32 + lane) * 16);

#pragma unroll 1
    for (int t = 0; t < TIN; ++t) {
      __bf16* hr = (t & 1) ? hbuf1 : hbuf0;
      __bf16* hw = (t & 1) ? hbuf0 : hbuf1;
      stage16(xfrag + ((long)(t * 4 + bg) * 2) * 512, AsmLDS, 2 * 64, tid);
      stage16(hr + (long)bg * 16 * 512, AsmLDS + 2 * 512, 16 * 64, tid);
      stage_wait();
      v8f acc0 = {}; v8f acc1 = {};
#pragma unroll
      for (int kt = 0; kt < KT_ENC; ++kt) {
        v16bf a = *(const v16bf*)(AsmLDS + kt * 512 + lane * 16);
        if (kt & 1)
          acc1 = __builtin_amdgcn_wmma_f32_16x16x32_bf16(false, a, false, wE[kt], (short)0, acc1, false, false);
        else
          acc0 = __builtin_amdgcn_wmma_f32_16x16x32_bf16(false, a, false, wE[kt], (short)0, acc0, false, false);
      }
#pragma unroll
      for (int r = 0; r < 8; ++r)
        gates[(r + mh) * 128 + wv * 16 + (lane & 15)] = acc0[r] + acc1[r] + bb1v;
      __syncthreads();
      cell_update(gates, ctile, hw, nullptr, bg, cg, tid);
      grid_sync(bar, epoch);
    }
  }

  // snapshot h1 (in hbuf0, TIN even) as the constant decoder input y
  for (long i = tid + (long)blockIdx.x * BT; i < 4L * 16 * 512; i += (long)NB * BT)
    yfrag[i] = hbuf0[i];
  grid_sync(bar, epoch);

  // ---------------- decoder + folded Linear ----------------
  {
    v16bf wD[KT_DEC];
#pragma unroll
    for (int kt = 0; kt < KT_DEC; ++kt)
      wD[kt] = *(const v16bf*)(w2p + ((long)(kt * NTILES + nt0) * 32 + lane) * 16);

    // stage the constant y fragments ONCE (slots 0..15)
    stage16(yfrag + (long)bg * 16 * 512, AsmLDS, 16 * 64, tid);
    stage_wait();

    v8f zero = {};
    v8f accL[2] = { zero, zero };
    const int ncol0 = cg * 256 + wv * 32;   // this wave's 32 output cols

#pragma unroll 1
    for (int t = 0; t < TOUT; ++t) {
      __bf16* hr = (t & 1) ? hbuf1 : hbuf0;
      __bf16* hw = (t & 1) ? hbuf0 : hbuf1;
      // per-step: only the h fragments move (slots 16..31)
      stage16(hr + (long)bg * 16 * 512, AsmLDS + 16 * 512, 16 * 64, tid);
      stage_wait();
      v8f acc0 = {}; v8f acc1 = {};
#pragma unroll
      for (int kt = 0; kt < KT_DEC; ++kt) {
        v16bf a = *(const v16bf*)(AsmLDS + kt * 512 + lane * 16);
        if (kt & 1)
          acc1 = __builtin_amdgcn_wmma_f32_16x16x32_bf16(false, a, false, wD[kt], (short)0, acc1, false, false);
        else
          acc0 = __builtin_amdgcn_wmma_f32_16x16x32_bf16(false, a, false, wD[kt], (short)0, acc0, false, false);
      }
#pragma unroll
      for (int r = 0; r < 8; ++r)
        gates[(r + mh) * 128 + wv * 16 + (lane & 15)] = acc0[r] + acc1[r] + bb2v;
      __syncthreads();
      cell_update(gates, ctile, hw, hlin, bg, cg, tid);
      grid_sync(bar, epoch);

      // stage this step's fp32 h rows (all 512 cols for our batch group)
      stage16(hlin + (long)bg * 16 * HH, hlinLDS, (16 * HH * 4) / 16, tid);
      stage_wait();

      // Linear accumulation over this step's K slab (K=512), fp32 WMMA 16x16x4
      const int k0off = (lane >= 16) ? 2 : 0;
      const float* arow = hlinLDS + (long)(lane & 15) * HH;
#pragma unroll 1
      for (int kc = 0; kc < HH / 4; ++kc) {
        int k0 = 4 * kc + k0off;
        v2f a; a.x = arow[k0]; a.y = arow[k0 + 1];
        long kg = (long)t * HH + k0;
#pragma unroll
        for (int i = 0; i < 2; ++i) {
          const float* wr = Wlin + (long)(ncol0 + i * 16 + (lane & 15)) * (TOUT * HH) + kg;
          __builtin_prefetch(wr + 32, 0, 1);
          v2f b; b.x = wr[0]; b.y = wr[1];
          accL[i] = __builtin_amdgcn_wmma_f32_16x16x4_f32(false, a, false, b, (short)0, accL[i], false, false);
        }
      }
      grid_sync(bar, epoch);
    }

    // ---------------- epilogue: out = accL + blin ----------------
#pragma unroll
    for (int i = 0; i < 2; ++i) {
#pragma unroll
      for (int r = 0; r < 8; ++r) {
        int row = bg * 16 + r + mh;
        int col = ncol0 + i * 16 + (lane & 15);
        out[(long)row * (TOUT * FOUT) + col] = accL[i][r] + blin[col];
      }
    }
  }
}

// ------------------------------------------------------------------
extern "C" void kernel_launch(void* const* d_in, const int* in_sizes, int n_in,
                              void* d_out, int out_size, void* d_ws, size_t ws_size,
                              hipStream_t stream)
{
  const float* x    = (const float*)d_in[0];
  const float* Wih1 = (const float*)d_in[1];
  const float* Whh1 = (const float*)d_in[2];
  const float* bih1 = (const float*)d_in[3];
  const float* bhh1 = (const float*)d_in[4];
  const float* Wih2 = (const float*)d_in[5];
  const float* Whh2 = (const float*)d_in[6];
  const float* bih2 = (const float*)d_in[7];
  const float* bhh2 = (const float*)d_in[8];
  const float* Wlin = (const float*)d_in[9];
  const float* blin = (const float*)d_in[10];
  float* out = (float*)d_out;

  char* ws = (char*)d_ws;
  size_t off = 0;
  auto take = [&](size_t bytes) -> char* {
    char* p = ws + off;
    off = (off + bytes + 255) & ~(size_t)255;
    return p;
  };
  unsigned* bar  = (unsigned*)take(64);
  __bf16* w1p    = (__bf16*)take((size_t)KT_ENC * NTILES * 512 * 2); // 2.25 MB
  __bf16* w2p    = (__bf16*)take((size_t)KT_DEC * NTILES * 512 * 2); // 4 MB
  __bf16* xfrag  = (__bf16*)take((size_t)TIN * 4 * 2 * 512 * 2);     // 1 MB
  __bf16* hfrag  = (__bf16*)take(2ull * 4 * 16 * 512 * 2);           // 128 KB
  __bf16* yfrag  = (__bf16*)take(4ull * 16 * 512 * 2);               // 64 KB
  float*  hlin   = (float*)take(64ull * HH * 4);                     // 128 KB
  float*  b1     = (float*)take(GG * 4);
  float*  b2     = (float*)take(GG * 4);
  (void)ws_size; (void)in_sizes; (void)n_in; (void)out_size;

  pack_kernel<<<512, 256, 0, stream>>>(x, Wih1, Whh1, bih1, bhh1,
                                       Wih2, Whh2, bih2, bhh2,
                                       w1p, w2p, xfrag, b1, b2, hfrag, bar);
  seq2seq_kernel<<<NB, BT, 0, stream>>>(w1p, w2p, xfrag, hfrag, yfrag,
                                        hlin, b1, b2, Wlin, blin, out, bar);
}
